// Center_pose_head_90194313216456
// MI455X (gfx1250) — matrix operand order repaired and checked by reading, not compiled, not run on hardware
//
#include <hip/hip_runtime.h>
#include <hip/hip_bf16.h>

typedef _Float16 h16;
typedef __attribute__((ext_vector_type(16))) _Float16 v16h;
typedef __attribute__((ext_vector_type(8)))  float    v8f;

union AFrag { uint4 q[2]; v16h v; };

static __device__ __forceinline__ int imin(int a, int b) { return a < b ? a : b; }
static __device__ __forceinline__ int imax(int a, int b) { return a > b ? a : b; }

// CDNA5 async DMA: global memory -> LDS, 16B per lane, tracked by ASYNCcnt.
// lds_off: byte offset into LDS (low 32 bits of generic pointer to __shared__).
static __device__ __forceinline__ void async_g2l_b128(unsigned lds_off, const void* g)
{
    asm volatile("global_load_async_to_lds_b128 %0, %1, off"
                 :: "v"(lds_off), "v"(g) : "memory");
}
static __device__ __forceinline__ void wait_async0()
{
    asm volatile("s_wait_asynccnt 0x0" ::: "memory");
}
static __device__ __forceinline__ unsigned lds_off32(const void* p)
{
    return (unsigned)(unsigned long long)p;   // LDS aperture: addr[31:0] = LDS offset
}

// ---------------------------------------------------------------------------
// Tiled WMMA GEMM:  out[m,n] = sum_k A[m,k] * B[n,k]   (both K-major, fp16)
// Block: 256 threads (8 wave32), tile M=64 x N=256, K step 32.
// Waves 2(M) x 4(N); each wave: 2x4 16x16 tiles -> 8 v_wmma per K-step.
// Double-buffered LDS, statically 2x-unrolled ping-pong: async DMA of tile
// k+1 overlaps WMMA compute of tile k, with no dynamic buffer indexing.
// Epilogue: v = v*scale[m] + shift[m]; optional ReLU.
//   mode 0: write fp16 NHWC  out[((b*OHo + y*sy+oy0)*OWo + x*sx+ox0)*OC + m]
//   mode 1: write fp32 NCHW  out[(b*Ctot + choff + m)*OHo*OWo + y*OWo + x]
// Requires: K % 32 == 0, N % 256 == 0, A padded to blockIdx.y*64+64 rows.
// ---------------------------------------------------------------------------
__global__ __launch_bounds__(256)
void gemm_f16_wmma(const h16* __restrict__ A, const h16* __restrict__ Bm,
                   int N, int K, int O,
                   const float* __restrict__ scale, const float* __restrict__ shift,
                   int relu, int mode,
                   h16* __restrict__ outh, float* __restrict__ outf,
                   int H, int W, int OC, int sy, int oy0, int sx, int ox0,
                   int OHo, int OWo, int choff, int Ctot)
{
    __shared__ uint4 sA[2][64 * 5];    // 64 rows x 32 f16, row stride 40 f16 (80B)
    __shared__ uint4 sB[2][256 * 5];   // 256 rows x 32 f16

    const int tid  = threadIdx.x;
    const int m0   = blockIdx.y * 64;
    const int n0   = blockIdx.x * 256;
    const int wave = tid >> 5;
    const int lane = tid & 31;
    const int wm   = wave & 1;        // 2 waves along M
    const int wn   = wave >> 1;       // 4 waves along N
    const int half = lane >> 4;
    const int l15  = lane & 15;
    const int r4   = tid >> 2;        // 0..63 : row for tile loads
    const int c4   = tid & 3;         // 0..3  : uint4 column
    const int Kq   = K >> 3;          // uint4 per row

    const uint4* __restrict__ Ag = (const uint4*)A;
    const uint4* __restrict__ Bg = (const uint4*)Bm;
    const size_t aBase = (size_t)(m0 + r4) * Kq + c4;
    size_t bBase[4];
#pragma unroll
    for (int j = 0; j < 4; ++j)
        bBase[j] = (size_t)(n0 + 64 * j + r4) * Kq + c4;

    unsigned ldsA[2], ldsB[2][4];
#pragma unroll
    for (int pb = 0; pb < 2; ++pb) {
        ldsA[pb] = lds_off32(&sA[pb][r4 * 5 + c4]);
#pragma unroll
        for (int j = 0; j < 4; ++j)
            ldsB[pb][j] = lds_off32(&sB[pb][(r4 + 64 * j) * 5 + c4]);
    }

    // Never executed (K > 0 always): keeps LDS arrays visibly written so the
    // compiler cannot fold the ds reads below (actual writes come from the
    // async DMA, which it cannot see).
    if (K == 0) {
        uint4 z = {0u, 0u, 0u, 0u};
        sA[0][tid] = z; sB[0][tid] = z;
        sA[1][tid] = z; sB[1][tid] = z;
    }

    const v8f vz = {0.f,0.f,0.f,0.f,0.f,0.f,0.f,0.f};
    v8f acc[2][4] = {{vz, vz, vz, vz}, {vz, vz, vz, vz}};

    const int ar0 = (wm * 32 + l15) * 5;
    const int ar1 = ar0 + 16 * 5;
    int br[4];
#pragma unroll
    for (int u = 0; u < 4; ++u)
        br[u] = (wn * 64 + u * 16 + l15) * 5;

    // issue async DMA of one K-tile into the buffer given by (lA, lB)
    auto issue = [&](unsigned lA, const unsigned* lB, int kqn, bool pf) {
        async_g2l_b128(lA, (const void*)(Ag + aBase + kqn));
#pragma unroll
        for (int j = 0; j < 4; ++j)
            async_g2l_b128(lB[j], (const void*)(Bg + bBase[j] + kqn));
        if (pf) {   // warm L2 one more tile ahead
            __builtin_prefetch((const void*)(Ag + aBase + kqn + 4), 0, 3);
            __builtin_prefetch((const void*)(Bg + bBase[0] + kqn + 4), 0, 3);
            __builtin_prefetch((const void*)(Bg + bBase[2] + kqn + 4), 0, 3);
        }
    };
    // 8 WMMAs on one staged K-tile
    auto compute = [&](const uint4* cA, const uint4* cB) {
        AFrag a0, a1, bf[4];
        a0.q[0] = cA[ar0 + half];      a0.q[1] = cA[ar0 + 2 + half];
        a1.q[0] = cA[ar1 + half];      a1.q[1] = cA[ar1 + 2 + half];
#pragma unroll
        for (int u = 0; u < 4; ++u) {
            bf[u].q[0] = cB[br[u] + half];
            bf[u].q[1] = cB[br[u] + 2 + half];
        }
#pragma unroll
        for (int u = 0; u < 4; ++u) {
            acc[0][u] = __builtin_amdgcn_wmma_f32_16x16x32_f16(false, a0.v, false, bf[u].v, (short)0, acc[0][u], false, false);
            acc[1][u] = __builtin_amdgcn_wmma_f32_16x16x32_f16(false, a1.v, false, bf[u].v, (short)0, acc[1][u], false, false);
        }
    };

    // ---- prologue: DMA tile 0 into buffer 0 ----
    issue(ldsA[0], ldsB[0], 0, false);
    wait_async0();
    __syncthreads();

    // ---- 2x-unrolled ping-pong pipeline (static buffer indices) ----
    int k0 = 0;
    for (;;) {
        // phase 0: compute buf0, DMA next tile into buf1
        if (k0 + 32 < K)
            issue(ldsA[1], ldsB[1], (k0 + 32) >> 3, k0 + 64 < K);
        compute(sA[0], sB[0]);
        wait_async0();
        __syncthreads();
        k0 += 32;
        if (k0 >= K) break;

        // phase 1: compute buf1, DMA next tile into buf0
        if (k0 + 32 < K)
            issue(ldsA[0], ldsB[0], (k0 + 32) >> 3, k0 + 64 < K);
        compute(sA[1], sB[1]);
        wait_async0();
        __syncthreads();
        k0 += 32;
        if (k0 >= K) break;
    }

    const int HW = H * W;
#pragma unroll
    for (int t = 0; t < 2; ++t)
#pragma unroll
    for (int u = 0; u < 4; ++u) {
        const int mb = m0 + wm * 32 + t * 16 + half * 8;
        const int n  = n0 + wn * 64 + u * 16 + l15;
        if (n >= N) continue;
        const int b   = n / HW;
        const int rem = n - b * HW;
        const int y   = rem / W;
        const int xx  = rem - y * W;
#pragma unroll
        for (int r = 0; r < 8; ++r) {
            const int m = mb + r;
            if (m >= O) continue;
            float v = acc[t][u][r];
            if (scale) v *= scale[m];
            if (shift) v += shift[m];
            if (relu)  v = fmaxf(v, 0.f);
            if (mode == 0) {
                const int oy = y * sy + oy0;
                const int ox = xx * sx + ox0;
                outh[((size_t)((b * OHo + oy) * OWo + ox)) * OC + m] = (h16)v;
            } else {
                outf[(size_t)(b * Ctot + choff + m) * (OHo * OWo) + (size_t)y * OWo + xx] = v;
            }
        }
    }
}

// ---------------------------------------------------------------------------
// Layout / packing kernels
// ---------------------------------------------------------------------------
__global__ void nchw_to_nhwc(const float* __restrict__ in, h16* __restrict__ out,
                             int B, int C, int H, int W)
{
    const size_t total = (size_t)B * C * H * W;
    for (size_t i = (size_t)blockIdx.x * blockDim.x + threadIdx.x; i < total;
         i += (size_t)gridDim.x * blockDim.x) {
        size_t r = i;
        const int c = (int)(r % C); r /= C;
        const int x = (int)(r % W); r /= W;
        const int y = (int)(r % H);
        const int b = (int)(r / H);
        out[i] = (h16)in[(((size_t)b * C + c) * H + y) * W + x];
    }
}

// w (O,C,3,3) fp32 -> wp [M64][9C] fp16, wp[o][k*C+c] = w[o][c][kh][kw], zero pad o>=O
__global__ void pack_conv3(const float* __restrict__ w, h16* __restrict__ wp,
                           int O, int C, int M64)
{
    const size_t K = 9 * (size_t)C;
    const size_t total = (size_t)M64 * K;
    for (size_t i = (size_t)blockIdx.x * blockDim.x + threadIdx.x; i < total;
         i += (size_t)gridDim.x * blockDim.x) {
        const int o = (int)(i / K);
        const size_t rem = i - (size_t)o * K;
        const int k = (int)(rem / C);
        const int c = (int)(rem - (size_t)k * C);
        float v = 0.f;
        if (o < O) v = w[((size_t)o * C + c) * 9 + k];
        wp[i] = (h16)v;
    }
}

// uw (I=C, O=C, 4, 4) -> wp[4 parities][C][4C]; ky = 3-a-2ty, kx = 3-b-2tx
__global__ void pack_deconv(const float* __restrict__ w, h16* __restrict__ wp, int C)
{
    const size_t K = 4 * (size_t)C;
    const size_t per = (size_t)C * K;
    const size_t total = 4 * per;
    for (size_t idx = (size_t)blockIdx.x * blockDim.x + threadIdx.x; idx < total;
         idx += (size_t)gridDim.x * blockDim.x) {
        const int p = (int)(idx / per);
        const size_t rem = idx - (size_t)p * per;
        const int o = (int)(rem / K);
        const size_t rem2 = rem - (size_t)o * K;
        const int t = (int)(rem2 / C);
        const int i = (int)(rem2 - (size_t)t * C);
        const int a = p >> 1, bb = p & 1;
        const int ty = t >> 1, tx = t & 1;
        const int ky = 3 - a - 2 * ty;
        const int kx = 3 - bb - 2 * tx;
        wp[idx] = (h16)w[(((size_t)i * C + o) * 4 + ky) * 4 + kx];
    }
}

__global__ void pack_1x1(const float* __restrict__ w, h16* __restrict__ wp,
                         int O, int C, int M64)
{
    const int total = M64 * C;
    for (int i = blockIdx.x * blockDim.x + threadIdx.x; i < total;
         i += gridDim.x * blockDim.x) {
        const int o = i / C, c = i - o * C;
        wp[i] = (h16)(o < O ? w[(size_t)o * C + c] : 0.f);
    }
}

__global__ void make_shift(const float* __restrict__ bias, const float* __restrict__ s,
                           const float* __restrict__ t, float* __restrict__ out, int n)
{
    const int i = blockIdx.x * blockDim.x + threadIdx.x;
    if (i < n) out[i] = bias[i] * s[i] + t[i];
}

// ---------------------------------------------------------------------------
// im2col: 3x3 pad 1, uint4 (8 x f16) chunked.  One block per output pixel.
// ---------------------------------------------------------------------------
__global__ void im2col3(const h16* __restrict__ x, h16* __restrict__ col,
                        int B, int H, int W, int C)
{
    const int row = blockIdx.x;
    const int HW = H * W;
    const int b = row / HW;
    const int rem = row - b * HW;
    const int y = rem / W;
    const int xx = rem - y * W;
    const int CH = C >> 3;
    const uint4* __restrict__ xs = (const uint4*)x;
    uint4* __restrict__ cs = (uint4*)col;
    const size_t ob = (size_t)row * 9 * CH;
    for (int idx = threadIdx.x; idx < 9 * CH; idx += blockDim.x) {
        const int k = idx / CH;
        const int cc = idx - k * CH;
        const int yy = y + k / 3 - 1;
        const int xc = xx + k % 3 - 1;
        uint4 v = {0u, 0u, 0u, 0u};
        if (yy >= 0 && yy < H && xc >= 0 && xc < W)
            v = xs[((size_t)((b * H + yy) * W + xc)) * CH + cc];
        cs[ob + idx] = v;
    }
}

// im2col for one parity sub-conv of deconv4s2p1: taps dy = ty-1+a, dx = tx-1+b
__global__ void im2col_dec(const h16* __restrict__ x, h16* __restrict__ col,
                           int B, int H, int W, int C, int pa, int pb)
{
    const int row = blockIdx.x;
    const int HW = H * W;
    const int b = row / HW;
    const int rem = row - b * HW;
    const int y = rem / W;
    const int xx = rem - y * W;
    const int CH = C >> 3;
    const uint4* __restrict__ xs = (const uint4*)x;
    uint4* __restrict__ cs = (uint4*)col;
    const size_t ob = (size_t)row * 4 * CH;
    for (int idx = threadIdx.x; idx < 4 * CH; idx += blockDim.x) {
        const int t = idx / CH;
        const int cc = idx - t * CH;
        const int yy = y + (t >> 1) - 1 + pa;
        const int xc = xx + (t & 1) - 1 + pb;
        uint4 v = {0u, 0u, 0u, 0u};
        if (yy >= 0 && yy < H && xc >= 0 && xc < W)
            v = xs[((size_t)((b * H + yy) * W + xc)) * CH + cc];
        cs[ob + idx] = v;
    }
}

// ---------------------------------------------------------------------------
// Deformable sampling: build masked bilinear im2col rows [BHW][9C] fp16.
// om rows [BHW][27] fp16: dy=om[2k], dx=om[2k+1], mask=sigmoid(om[18+k]).
// ---------------------------------------------------------------------------
__global__ void dcn_sample(const h16* __restrict__ x, const h16* __restrict__ om,
                           h16* __restrict__ col, int B, int H, int W, int C)
{
    const int row = blockIdx.x;
    const int HW = H * W;
    const int b = row / HW;
    const int rem = row - b * HW;
    const int y = rem / W;
    const int xx = rem - y * W;
    const h16* __restrict__ omr = om + (size_t)row * 27;
    const size_t xb = (size_t)b * HW * C;
    const size_t ob = (size_t)row * 9 * C;
    for (int k = 0; k < 9; ++k) {
        const float dy = (float)omr[2 * k];
        const float dx = (float)omr[2 * k + 1];
        const float mk = 1.0f / (1.0f + __expf(-(float)omr[18 + k]));
        const float py = (float)y + (float)(k / 3 - 1) + dy;
        const float px = (float)xx + (float)(k % 3 - 1) + dx;
        const float y0 = floorf(py), x0 = floorf(px);
        const float wy = py - y0, wx = px - x0;
        const bool vy0 = (y0 >= 0.f) && (y0 <= (float)(H - 1));
        const bool vy1 = (y0 + 1.f >= 0.f) && (y0 + 1.f <= (float)(H - 1));
        const bool vx0 = (x0 >= 0.f) && (x0 <= (float)(W - 1));
        const bool vx1 = (x0 + 1.f >= 0.f) && (x0 + 1.f <= (float)(W - 1));
        const int cy0 = imin(imax((int)y0, 0), H - 1);
        const int cy1 = imin(imax((int)y0 + 1, 0), H - 1);
        const int cx0 = imin(imax((int)x0, 0), W - 1);
        const int cx1 = imin(imax((int)x0 + 1, 0), W - 1);
        const float w00 = (vy0 && vx0) ? (1.f - wy) * (1.f - wx) : 0.f;
        const float w01 = (vy0 && vx1) ? (1.f - wy) * wx : 0.f;
        const float w10 = (vy1 && vx0) ? wy * (1.f - wx) : 0.f;
        const float w11 = (vy1 && vx1) ? wy * wx : 0.f;
        const size_t b00 = xb + ((size_t)cy0 * W + cx0) * C;
        const size_t b01 = xb + ((size_t)cy0 * W + cx1) * C;
        const size_t b10 = xb + ((size_t)cy1 * W + cx0) * C;
        const size_t b11 = xb + ((size_t)cy1 * W + cx1) * C;
        for (int c = threadIdx.x; c < C; c += blockDim.x) {
            const float s = w00 * (float)x[b00 + c] + w01 * (float)x[b01 + c] +
                            w10 * (float)x[b10 + c] + w11 * (float)x[b11 + c];
            col[ob + (size_t)k * C + c] = (h16)(s * mk);
        }
    }
}

// ---------------------------------------------------------------------------
// Host side
// ---------------------------------------------------------------------------
static void run_gemm(hipStream_t stream, const h16* A, const h16* Bm, int N, int K,
                     int O, int M64, const float* scale, const float* shift, int relu,
                     int mode, h16* outh, float* outf, int H, int W, int OC,
                     int sy, int oy0, int sx, int ox0, int OHo, int OWo,
                     int choff, int Ctot)
{
    dim3 g(N / 256, M64 / 64);
    gemm_f16_wmma<<<g, 256, 0, stream>>>(A, Bm, N, K, O, scale, shift, relu, mode,
                                         outh, outf, H, W, OC, sy, oy0, sx, ox0,
                                         OHo, OWo, choff, Ctot);
}

extern "C" void kernel_launch(void* const* d_in, const int* in_sizes, int n_in,
                              void* d_out, int out_size, void* d_ws, size_t ws_size,
                              hipStream_t stream)
{
    (void)in_sizes; (void)n_in; (void)out_size; (void)ws_size;

    const float* xin    = (const float*)d_in[0];
    const float* dwo[3] = {(const float*)d_in[1],  (const float*)d_in[10], (const float*)d_in[19]};
    const float* dbo[3] = {(const float*)d_in[2],  (const float*)d_in[11], (const float*)d_in[20]};
    const float* dwm[3] = {(const float*)d_in[3],  (const float*)d_in[12], (const float*)d_in[21]};
    const float* dbm[3] = {(const float*)d_in[4],  (const float*)d_in[13], (const float*)d_in[22]};
    const float* s1v[3] = {(const float*)d_in[5],  (const float*)d_in[14], (const float*)d_in[23]};
    const float* t1v[3] = {(const float*)d_in[6],  (const float*)d_in[15], (const float*)d_in[24]};
    const float* uwv[3] = {(const float*)d_in[7],  (const float*)d_in[16], (const float*)d_in[25]};
    const float* s2v[3] = {(const float*)d_in[8],  (const float*)d_in[17], (const float*)d_in[26]};
    const float* t2v[3] = {(const float*)d_in[9],  (const float*)d_in[18], (const float*)d_in[27]};
    const float* hw1[3] = {(const float*)d_in[28], (const float*)d_in[32], (const float*)d_in[36]};
    const float* hb1[3] = {(const float*)d_in[29], (const float*)d_in[33], (const float*)d_in[37]};
    const float* hw2[3] = {(const float*)d_in[30], (const float*)d_in[34], (const float*)d_in[38]};
    const float* hb2[3] = {(const float*)d_in[31], (const float*)d_in[35], (const float*)d_in[39]};

    const int Bn = 4;
    const int Ci[3] = {1024, 512, 256};
    const int Co[3] = {512, 256, 64};
    const int Hi[3] = {16, 32, 64};
    const int cls[3] = {34, 17, 2};
    const int choff[3] = {0, 34, 51};

    // ---- workspace layout ----
    char* base = (char*)d_ws;
    size_t off = 0;
    auto A16 = [&](size_t elems) -> h16* {
        h16* r = (h16*)(base + off);
        off = (off + elems * 2 + 255) & ~(size_t)255;
        return r;
    };
    auto A32 = [&](size_t elems) -> float* {
        float* r = (float*)(base + off);
        off = (off + elems * 4 + 255) & ~(size_t)255;
        return r;
    };
    h16* WPO[3]; h16* WDW[3]; h16* WUW[3]; h16* WH1[3]; h16* WH2[3]; float* SHB[3];
    for (int i = 0; i < 3; ++i) WPO[i] = A16((size_t)64 * 9 * Ci[i]);
    for (int i = 0; i < 3; ++i) WDW[i] = A16((size_t)Co[i] * 9 * Ci[i]);
    for (int i = 0; i < 3; ++i) WUW[i] = A16((size_t)4 * Co[i] * 4 * Co[i]);
    for (int i = 0; i < 3; ++i) WH1[i] = A16((size_t)256 * 576);
    for (int i = 0; i < 3; ++i) WH2[i] = A16((size_t)64 * 256);
    for (int i = 0; i < 3; ++i) SHB[i] = A32((size_t)Co[i]);
    h16* OM   = A16((size_t)16384 * 27);
    h16* COL  = A16((size_t)65536 * 576);          // 37.75M f16, reused everywhere
    h16* ACT0 = A16((size_t)4 * 128 * 128 * 64);   // ping
    h16* ACT1 = A16((size_t)4 * 64 * 64 * 64 * 4); // pong (dcn outs)
    h16* ACT2 = A16((size_t)4 * 128 * 128 * 256);  // head hidden

    // ---- weight packing / affine fusion ----
    nchw_to_nhwc<<<2048, 256, 0, stream>>>(xin, ACT0, Bn, 1024, 16, 16);
    for (int i = 0; i < 3; ++i) {
        pack_conv3<<<1024, 256, 0, stream>>>(dwo[i], WPO[i], 27, Ci[i], 64);
        pack_conv3<<<2048, 256, 0, stream>>>(dwm[i], WDW[i], Co[i], Ci[i], Co[i]);
        pack_deconv<<<2048, 256, 0, stream>>>(uwv[i], WUW[i], Co[i]);
        make_shift<<<(Co[i] + 255) / 256, 256, 0, stream>>>(dbm[i], s1v[i], t1v[i], SHB[i], Co[i]);
    }
    for (int hh = 0; hh < 3; ++hh) {
        pack_conv3<<<1024, 256, 0, stream>>>(hw1[hh], WH1[hh], 256, 64, 256);
        pack_1x1<<<64, 256, 0, stream>>>(hw2[hh], WH2[hh], cls[hh], 256, 64);
    }

    // ---- backbone: 3 x (DCN -> BN/ReLU -> deconv4s2p1 -> BN/ReLU) ----
    for (int i = 0; i < 3; ++i) {
        const int C = Ci[i], O = Co[i], H = Hi[i], W = Hi[i];
        const int N = Bn * H * W;
        // offset/mask conv (27 ch, bias only)
        im2col3<<<N, 256, 0, stream>>>(ACT0, COL, Bn, H, W, C);
        run_gemm(stream, WPO[i], COL, N, 9 * C, 27, 64, nullptr, dbo[i], 0,
                 0, OM, nullptr, H, W, 27, 1, 0, 1, 0, H, W, 0, 0);
        // deformable bilinear im2col
        dcn_sample<<<N, 256, 0, stream>>>(ACT0, OM, COL, Bn, H, W, C);
        // main DCN conv + fused BN/ReLU  (v*s1 + (db*s1 + t1))
        run_gemm(stream, WDW[i], COL, N, 9 * C, O, O, s1v[i], SHB[i], 1,
                 0, ACT1, nullptr, H, W, O, 1, 0, 1, 0, H, W, 0, 0);
        // deconv 4x4 s2 p1 as 4 parity sub-convs + fused BN/ReLU
        for (int p = 0; p < 4; ++p) {
            const int pa = p >> 1, pb = p & 1;
            im2col_dec<<<N, 256, 0, stream>>>(ACT1, COL, Bn, H, W, O, pa, pb);
            run_gemm(stream, WUW[i] + (size_t)p * O * 4 * O, COL, N, 4 * O, O, O,
                     s2v[i], t2v[i], 1, 0, ACT0, nullptr, H, W, O,
                     2, pa, 2, pb, 2 * H, 2 * W, 0, 0);
        }
    }

    // ---- heads: shared 3x3 im2col of h, then per-head 3x3+ReLU and 1x1 ----
    const int Nh = Bn * 128 * 128;
    im2col3<<<Nh, 256, 0, stream>>>(ACT0, COL, Bn, 128, 128, 64);
    for (int hh = 0; hh < 3; ++hh) {
        run_gemm(stream, WH1[hh], COL, Nh, 576, 256, 256, nullptr, hb1[hh], 1,
                 0, ACT2, nullptr, 128, 128, 256, 1, 0, 1, 0, 128, 128, 0, 0);
        run_gemm(stream, WH2[hh], ACT2, Nh, 256, cls[hh], 64, nullptr, hb2[hh], 0,
                 1, nullptr, (float*)d_out, 128, 128, 0, 1, 0, 1, 0, 128, 128,
                 choff[hh], 53);
    }
}
